// TransformerEncoderLayer_35536559407328
// MI455X (gfx1250) — matrix operand order repaired and checked by reading, not compiled, not run on hardware
//
#include <hip/hip_runtime.h>

#define DEVINL __device__ __forceinline__
#define USE_TDM 1

typedef __attribute__((ext_vector_type(16))) __bf16 v16bf;
typedef __attribute__((ext_vector_type(8)))  __bf16 v8bf;
typedef __attribute__((ext_vector_type(8)))  float  v8f;
typedef __attribute__((ext_vector_type(4)))  unsigned int u32x4;
typedef __attribute__((ext_vector_type(8)))  unsigned int u32x8;

union V16 { v16bf v; v8bf h[2]; };

static constexpr int T_TOK = 4096;   // 4 * 1024 tokens
static constexpr int DMODEL = 512;
static constexpr int NEXP = 32;
static constexpr int TOPK = 4;

DEVINL __bf16 f2bf(float f) { return (__bf16)f; }
DEVINL float  bf2f(__bf16 b) { return (float)b; }

DEVINL v8bf cvt8(float4 a, float4 b) {
  v8bf r;
  r[0] = (__bf16)a.x; r[1] = (__bf16)a.y; r[2] = (__bf16)a.z; r[3] = (__bf16)a.w;
  r[4] = (__bf16)b.x; r[5] = (__bf16)b.y; r[6] = (__bf16)b.z; r[7] = (__bf16)b.w;
  return r;
}

DEVINL float gelu_t(float x) {
  return 0.5f * x * (1.0f + tanhf(0.7978845608028654f * (x + 0.044715f * x * x * x)));
}

DEVINL v8f wmma_bf16(v16bf a, v16bf b, v8f c) {
  return __builtin_amdgcn_wmma_f32_16x16x32_bf16(false, a, false, b, (short)0, c,
                                                 false, false);
}

// gfx1250 async global->LDS copy (GLOBAL_LOAD_ASYNC_TO_LDS_B128, ASYNCcnt).
// LDS operand = low 32 bits of the generic address (== LDS byte offset, ISA 10.2).
DEVINL void async_b128(const void* g, void* l) {
  asm volatile("global_load_async_to_lds_b128 %0, %1, off"
               :: "v"((unsigned)(unsigned long long)l),
                  "v"((unsigned long long)g)
               : "memory");
}

#if __has_builtin(__builtin_amdgcn_s_wait_asynccnt)
#define WAIT_ASYNC(n) __builtin_amdgcn_s_wait_asynccnt(n)
#else
#define WAIT_ASYNC(n) asm volatile("s_wait_asynccnt %0" :: "n"(n) : "memory")
#endif

#if __has_builtin(__builtin_amdgcn_s_wait_tensorcnt)
#define WAIT_TENSOR(n) __builtin_amdgcn_s_wait_tensorcnt(n)
#else
#define WAIT_TENSOR(n) asm volatile("s_wait_tensorcnt %0" :: "n"(n) : "memory")
#endif

#if USE_TDM
// TDM: load one 32(row) x 64(col) bf16 tile from a row-major [K,N] matrix into
// LDS rows padded to 144B (64 elems + 8 pad). D# per ISA 08_async_tensor 8.3/8.4:
//   pad_interval code 4 -> pad after 32 DWORDs (one 128B row)
//   pad_amount  code 3 -> 4 DWORDs (16B) padding  => 144B LDS row stride
DEVINL void tdm_load_b_tile(const void* gsrc, void* ldst, unsigned n_stride) {
  unsigned long long ga = (unsigned long long)gsrc;
  unsigned la = (unsigned)(unsigned long long)ldst;
  u32x4 g0;
  g0[0] = 1u;                                        // count=1 (user mode)
  g0[1] = la;                                        // lds_addr
  g0[2] = (unsigned)ga;                              // global_addr[31:0]
  g0[3] = ((unsigned)(ga >> 32) & 0x01FFFFFFu) | (2u << 30);  // addr[56:32]|type=2
  u32x8 g1;
  g1[0] = (1u << 16) | (1u << 20) | (4u << 22) | (3u << 25);  // 2B, pad on, 32dw/4dw
  g1[1] = (n_stride & 0xFFFFu) << 16;                // tensor_dim0[15:0]
  g1[2] = ((n_stride >> 16) & 0xFFFFu) | (32u << 16);// dim0[31:16], dim1=32
  g1[3] = (64u << 16);                               // dim1 hi=0, tile_dim0=64
  g1[4] = 32u;                                       // tile_dim1=32, tile_dim2=0
  g1[5] = n_stride;                                  // dim0_stride[31:0] (elems)
  g1[6] = 0u;                                        // stride[47:32]=0, d1stride=0
  g1[7] = 0u;
  asm volatile("tensor_load_to_lds %0, %1" :: "s"(g0), "s"(g1) : "memory");
}
#endif

// ---------------------------------------------------------------------------
// fp32 -> bf16 bulk conversion (weights; n multiple of 8)
// ---------------------------------------------------------------------------
__global__ __launch_bounds__(256)
void cvt_bf16_kernel(const float* __restrict__ in, __bf16* __restrict__ out, int n) {
  int i = (blockIdx.x * 256 + threadIdx.x) * 8;
  if (i >= n) return;
  float4 a = *reinterpret_cast<const float4*>(in + i);
  float4 b = *reinterpret_cast<const float4*>(in + i + 4);
  *reinterpret_cast<v8bf*>(out + i) = cvt8(a, b);
}

// ---------------------------------------------------------------------------
// LayerNorm: fp32 [T,512] -> bf16 [T,512]
// ---------------------------------------------------------------------------
__global__ __launch_bounds__(256)
void ln_kernel(const float* __restrict__ x, const float* __restrict__ g,
               const float* __restrict__ b, __bf16* __restrict__ out) {
  int t = blockIdx.x;
  int tid = threadIdx.x;
  const float* xr = x + (size_t)t * DMODEL;
  float a0 = xr[tid], a1 = xr[tid + 256];
  float s = a0 + a1, q = a0 * a0 + a1 * a1;
#pragma unroll
  for (int off = 16; off >= 1; off >>= 1) {
    s += __shfl_xor(s, off, 32);
    q += __shfl_xor(q, off, 32);
  }
  __shared__ float red[16];
  __shared__ float mv[2];
  int wid = tid >> 5, lane = tid & 31;
  if (lane == 0) { red[wid] = s; red[wid + 8] = q; }
  __syncthreads();
  if (tid == 0) {
    float ts = 0.f, tq = 0.f;
    for (int i = 0; i < 8; ++i) { ts += red[i]; tq += red[i + 8]; }
    float mu = ts * (1.0f / DMODEL);
    float var = tq * (1.0f / DMODEL) - mu * mu;
    mv[0] = mu; mv[1] = rsqrtf(var + 1e-5f);
  }
  __syncthreads();
  float mu = mv[0], inv = mv[1];
  out[(size_t)t * DMODEL + tid] = f2bf((a0 - mu) * inv * g[tid] + b[tid]);
  out[(size_t)t * DMODEL + tid + 256] =
      f2bf((a1 - mu) * inv * g[tid + 256] + b[tid + 256]);
}

// ---------------------------------------------------------------------------
// Dense WMMA GEMM; A staged by per-lane async copies, B staged by TDM (wave 0).
// C[M,N] = A[M,K](bf16) * B[K,N](bf16) + bias
// Block tile 128x64, wave tile 32x32 (8 waves).
// EPI: 0 = store bf16, 1 = kT scatter bf16, 2 = gelu + bf16,
//      3 = +residual, store f32, 4 = store f32
// ---------------------------------------------------------------------------
template <int EPI>
__global__ __launch_bounds__(256)
void gemm_dense(const __bf16* __restrict__ A, const __bf16* __restrict__ B,
                const float* __restrict__ bias, const float* __restrict__ resid,
                __bf16* __restrict__ outB, float* __restrict__ outF,
                int M, int N, int K) {
  __shared__ __bf16 As[2][128][40];
  __shared__ __bf16 Bs[2][32][72];
  const int tid = threadIdx.x;
  const int mrow0 = blockIdx.x * 128;
  const int n0g = blockIdx.y * 64;
  const int lane = tid & 31;
  const int wid = tid >> 5;

  const __bf16* arow = A + (size_t)(mrow0 + (tid >> 1)) * K + (tid & 1) * 16;
  const __bf16* bbase = B + n0g;

  auto stageA = [&](int kt, int buf) {
    int k0 = kt * 32;
    async_b128(arow + k0, &As[buf][tid >> 1][(tid & 1) * 16]);
    async_b128(arow + k0 + 8, &As[buf][tid >> 1][(tid & 1) * 16 + 8]);
  };
#if USE_TDM
  auto stageB = [&](int kt, int buf) {
    if (wid == 0) tdm_load_b_tile(bbase + (size_t)kt * 32 * N, &Bs[buf][0][0], N);
  };
#else
  const __bf16* brow = B + (size_t)(tid >> 3) * N + n0g + (tid & 7) * 8;
  auto stageB = [&](int kt, int buf) {
    async_b128(brow + (size_t)kt * 32 * N, &Bs[buf][tid >> 3][(tid & 7) * 8]);
  };
#endif

  const int wm = wid & 3;   // 32-row block
  const int wn = wid >> 2;  // 32-col block

  v8f acc[2][2];
#pragma unroll
  for (int a = 0; a < 2; ++a)
#pragma unroll
    for (int c = 0; c < 2; ++c) acc[a][c] = (v8f){0, 0, 0, 0, 0, 0, 0, 0};

  const int akb = (lane >> 4) * 8;
  const int nk = K / 32;

  stageA(0, 0);
  stageB(0, 0);
  for (int it = 0; it < nk; ++it) {
    const int cur = it & 1;
    if (it + 1 < nk) {
      stageA(it + 1, 1 - cur);
      stageB(it + 1, 1 - cur);
      WAIT_ASYNC(2);
#if USE_TDM
      if (wid == 0) WAIT_TENSOR(1);
#else
      WAIT_ASYNC(2);  // (unreachable split kept trivial)
#endif
    } else {
      WAIT_ASYNC(0);
#if USE_TDM
      if (wid == 0) WAIT_TENSOR(0);
#endif
    }
    __syncthreads();

    V16 af[2], bf[2];
#pragma unroll
    for (int a = 0; a < 2; ++a) {
      int am = wm * 32 + a * 16 + (lane & 15);
      af[a].h[0] = *reinterpret_cast<const v8bf*>(&As[cur][am][akb]);
      af[a].h[1] = *reinterpret_cast<const v8bf*>(&As[cur][am][akb + 16]);
    }
#pragma unroll
    for (int c = 0; c < 2; ++c) {
#pragma unroll
      for (int s = 0; s < 16; ++s) {
        int kk = (s < 8) ? akb + s : 16 + akb + (s - 8);
        bf[c].v[s] = Bs[cur][kk][wn * 32 + c * 16 + (lane & 15)];
      }
    }
#pragma unroll
    for (int a = 0; a < 2; ++a)
#pragma unroll
      for (int c = 0; c < 2; ++c) acc[a][c] = wmma_bf16(af[a].v, bf[c].v, acc[a][c]);
    __syncthreads();
  }

#pragma unroll
  for (int a = 0; a < 2; ++a)
#pragma unroll
    for (int c = 0; c < 2; ++c)
#pragma unroll
      for (int vv = 0; vv < 8; ++vv) {
        int grow = mrow0 + wm * 32 + a * 16 + vv + ((lane >> 4) << 3);
        int gcol = n0g + wn * 32 + c * 16 + (lane & 15);
        float val = acc[a][c][vv] + bias[gcol];
        if constexpr (EPI == 2) val = gelu_t(val);
        if constexpr (EPI == 3) val += resid[(size_t)grow * N + gcol];
        if constexpr (EPI == 0 || EPI == 2) {
          outB[(size_t)grow * N + gcol] = f2bf(val);
        } else if constexpr (EPI == 1) {
          int bb = grow >> 10, ss = grow & 1023;
          int hh = gcol >> 6, dd = gcol & 63;
          outB[(size_t)((bb * 8 + hh) * 64 + dd) * 1024 + ss] = f2bf(val);
        } else {
          outF[(size_t)grow * N + gcol] = val;
        }
      }
}

// ---------------------------------------------------------------------------
// Per-expert WMMA GEMM (same engine). FIRST: gather rows via token list,
// gelu, bf16 out.  !FIRST: contiguous slot rows, fp32 out.
// Rows beyond the expert's count load arbitrary valid data and are not stored.
// ---------------------------------------------------------------------------
template <bool FIRST>
__global__ __launch_bounds__(256)
void gemm_expert(const __bf16* __restrict__ A, const __bf16* __restrict__ Wb,
                 const float* __restrict__ Bb, __bf16* __restrict__ outB,
                 float* __restrict__ outF, const int* __restrict__ token_list,
                 const int* __restrict__ offsets) {
  const int e = blockIdx.z;
  const int start = offsets[e], end = offsets[e + 1];
  const int cnt = end - start;
  const int mb = blockIdx.x;
  if (mb * 128 >= cnt) return;

  __shared__ __bf16 As[2][128][40];
  __shared__ __bf16 Bs[2][32][72];
  const __bf16* W = Wb + (size_t)e * DMODEL * DMODEL;
  const float* bias = Bb + (size_t)e * DMODEL;

  const int tid = threadIdx.x;
  const int n0g = blockIdx.y * 64;
  const int lane = tid & 31;
  const int wid = tid >> 5;

  int slot = start + mb * 128 + (tid >> 1);
  if (slot >= end) slot = end - 1;  // clamp: garbage rows are never stored
  const __bf16* arow;
  if constexpr (FIRST) {
    arow = A + (size_t)token_list[slot] * DMODEL + (tid & 1) * 16;
  } else {
    arow = A + (size_t)slot * DMODEL + (tid & 1) * 16;
  }
  const __bf16* bbase = W + n0g;

  auto stageA = [&](int kt, int buf) {
    int k0 = kt * 32;
    async_b128(arow + k0, &As[buf][tid >> 1][(tid & 1) * 16]);
    async_b128(arow + k0 + 8, &As[buf][tid >> 1][(tid & 1) * 16 + 8]);
  };
#if USE_TDM
  auto stageB = [&](int kt, int buf) {
    if (wid == 0)
      tdm_load_b_tile(bbase + (size_t)kt * 32 * DMODEL, &Bs[buf][0][0], DMODEL);
  };
#else
  const __bf16* brow = W + (size_t)(tid >> 3) * DMODEL + n0g + (tid & 7) * 8;
  auto stageB = [&](int kt, int buf) {
    async_b128(brow + (size_t)kt * 32 * DMODEL, &Bs[buf][tid >> 3][(tid & 7) * 8]);
  };
#endif

  const int wm = wid & 3, wn = wid >> 2;
  v8f acc[2][2];
#pragma unroll
  for (int a = 0; a < 2; ++a)
#pragma unroll
    for (int c = 0; c < 2; ++c) acc[a][c] = (v8f){0, 0, 0, 0, 0, 0, 0, 0};
  const int akb = (lane >> 4) * 8;
  const int nk = DMODEL / 32;

  stageA(0, 0);
  stageB(0, 0);
  for (int it = 0; it < nk; ++it) {
    const int cur = it & 1;
    if (it + 1 < nk) {
      stageA(it + 1, 1 - cur);
      stageB(it + 1, 1 - cur);
      WAIT_ASYNC(2);
#if USE_TDM
      if (wid == 0) WAIT_TENSOR(1);
#endif
    } else {
      WAIT_ASYNC(0);
#if USE_TDM
      if (wid == 0) WAIT_TENSOR(0);
#endif
    }
    __syncthreads();

    V16 af[2], bf[2];
#pragma unroll
    for (int a = 0; a < 2; ++a) {
      int am = wm * 32 + a * 16 + (lane & 15);
      af[a].h[0] = *reinterpret_cast<const v8bf*>(&As[cur][am][akb]);
      af[a].h[1] = *reinterpret_cast<const v8bf*>(&As[cur][am][akb + 16]);
    }
#pragma unroll
    for (int c = 0; c < 2; ++c) {
#pragma unroll
      for (int s = 0; s < 16; ++s) {
        int kk = (s < 8) ? akb + s : 16 + akb + (s - 8);
        bf[c].v[s] = Bs[cur][kk][wn * 32 + c * 16 + (lane & 15)];
      }
    }
#pragma unroll
    for (int a = 0; a < 2; ++a)
#pragma unroll
      for (int c = 0; c < 2; ++c) acc[a][c] = wmma_bf16(af[a].v, bf[c].v, acc[a][c]);
    __syncthreads();
  }

#pragma unroll
  for (int a = 0; a < 2; ++a)
#pragma unroll
    for (int c = 0; c < 2; ++c)
#pragma unroll
      for (int vv = 0; vv < 8; ++vv) {
        int oslot = start + mb * 128 + wm * 32 + a * 16 + vv + ((lane >> 4) << 3);
        if (oslot < end) {
          int gcol = n0g + wn * 32 + c * 16 + (lane & 15);
          float val = acc[a][c][vv] + bias[gcol];
          if constexpr (FIRST) {
            outB[(size_t)oslot * DMODEL + gcol] = f2bf(gelu_t(val));
          } else {
            outF[(size_t)oslot * DMODEL + gcol] = val;
          }
        }
      }
}

// ---------------------------------------------------------------------------
// Flash attention: 1 wave per (b,h,16-query block); online softmax, bf16 WMMA
// q: [T,512] bf16; kT: [B,H,64,1024] bf16; v: [T,512] bf16; o: [T,512] bf16
// ---------------------------------------------------------------------------
__global__ __launch_bounds__(32)
void attn_kernel(const __bf16* __restrict__ q, const __bf16* __restrict__ kT,
                 const __bf16* __restrict__ v, __bf16* __restrict__ o) {
  __shared__ __bf16 Pl[16][40];
  const int gid = blockIdx.x;
  const int qb = gid & 63;
  const int bh = gid >> 6;
  const int b = bh >> 3;
  const int lane = threadIdx.x;
  const int row0 = b * 1024 + qb * 16;
  const int colh = (bh & 7) * 64;

  const int m = lane & 15;
  const int kb = (lane >> 4) * 8;

  V16 qf[2];
#pragma unroll
  for (int c = 0; c < 2; ++c) {
    const __bf16* p = q + (size_t)(row0 + m) * DMODEL + colh + c * 32 + kb;
    qf[c].h[0] = *reinterpret_cast<const v8bf*>(p);
    qf[c].h[1] = *reinterpret_cast<const v8bf*>(p + 16);
  }

  float mstate[8], lstate[8];
#pragma unroll
  for (int i = 0; i < 8; ++i) { mstate[i] = -1e30f; lstate[i] = 0.f; }
  v8f zf = {0, 0, 0, 0, 0, 0, 0, 0};
  v8f oacc[4];
#pragma unroll
  for (int d = 0; d < 4; ++d) oacc[d] = zf;

  for (int j = 0; j < 32; ++j) {
    const int key0 = j * 32;
    v8f sc[2];
#pragma unroll
    for (int half = 0; half < 2; ++half) {
      v8f s = zf;
#pragma unroll
      for (int c = 0; c < 2; ++c) {
        const __bf16* kp =
            kT + (size_t)(bh * 64 + c * 32) * 1024 + key0 + half * 16 + m;
        if (half == 0 && j + 1 < 32) __builtin_prefetch(kp + 32, 0, 1);
        V16 kf;
#pragma unroll
        for (int s = 0; s < 16; ++s) {
          int kk = (s < 8) ? kb + s : 16 + kb + (s - 8);
          kf.v[s] = kp[(size_t)kk * 1024];
        }
        s = wmma_bf16(qf[c].v, kf.v, s);
      }
      sc[half] = s;
    }
    float corr[8];
#pragma unroll
    for (int vv = 0; vv < 8; ++vv) {
      float s0 = sc[0][vv] * 0.125f;
      float s1 = sc[1][vv] * 0.125f;
      float r = fmaxf(s0, s1);
#pragma unroll
      for (int off = 1; off <= 8; off <<= 1) r = fmaxf(r, __shfl_xor(r, off, 32));
      float nm = fmaxf(mstate[vv], r);
      float p0 = __expf(s0 - nm);
      float p1 = __expf(s1 - nm);
      float rs = p0 + p1;
#pragma unroll
      for (int off = 1; off <= 8; off <<= 1) rs += __shfl_xor(rs, off, 32);
      float cr = __expf(mstate[vv] - nm);
      lstate[vv] = lstate[vv] * cr + rs;
      mstate[vv] = nm;
      corr[vv] = cr;
      int qrow = vv + ((lane >> 4) << 3);
      int kcol = lane & 15;
      Pl[qrow][kcol] = f2bf(p0);
      Pl[qrow][kcol + 16] = f2bf(p1);
    }
#pragma unroll
    for (int d = 0; d < 4; ++d)
#pragma unroll
      for (int vv = 0; vv < 8; ++vv) oacc[d][vv] *= corr[vv];

    V16 pf;
    pf.h[0] = *reinterpret_cast<const v8bf*>(&Pl[m][kb]);
    pf.h[1] = *reinterpret_cast<const v8bf*>(&Pl[m][kb + 16]);
#pragma unroll
    for (int d = 0; d < 4; ++d) {
      const __bf16* vp =
          v + (size_t)(b * 1024 + key0) * DMODEL + colh + d * 16 + m;
      if (d == 0 && j + 1 < 32) __builtin_prefetch(vp + 32 * DMODEL, 0, 1);
      V16 vf;
#pragma unroll
      for (int s = 0; s < 16; ++s) {
        int kk = (s < 8) ? kb + s : 16 + kb + (s - 8);
        vf.v[s] = vp[(size_t)kk * DMODEL];
      }
      oacc[d] = wmma_bf16(pf.v, vf.v, oacc[d]);
    }
  }
#pragma unroll
  for (int d = 0; d < 4; ++d)
#pragma unroll
    for (int vv = 0; vv < 8; ++vv) {
      int row = row0 + vv + ((lane >> 4) << 3);
      int col = colh + d * 16 + (lane & 15);
      o[(size_t)row * DMODEL + col] = f2bf(oacc[d][vv] / lstate[vv]);
    }
}

// ---------------------------------------------------------------------------
// Low-rank router + softmax + top-4 per token (one block per token)
// ---------------------------------------------------------------------------
__global__ __launch_bounds__(64)
void router_kernel(const __bf16* __restrict__ h2, const float* __restrict__ ra,
                   const float* __restrict__ rb, float* __restrict__ probs,
                   int* __restrict__ topi, float* __restrict__ topw) {
  int t = blockIdx.x;
  int tid = threadIdx.x;
  __shared__ float mid[64];
  const __bf16* hr = h2 + (size_t)t * DMODEL;
  float acc = 0.f;
  for (int d = 0; d < DMODEL; ++d) acc += bf2f(hr[d]) * ra[d * 64 + tid];
  mid[tid] = acc;
  __syncthreads();
  if (tid < 32) {
    float lg = 0.f;
    for (int r = 0; r < 64; ++r) lg += mid[r] * rb[r * NEXP + tid];
    float mx = lg;
#pragma unroll
    for (int off = 16; off >= 1; off >>= 1) mx = fmaxf(mx, __shfl_xor(mx, off, 32));
    float p = __expf(lg - mx);
    float sm = p;
#pragma unroll
    for (int off = 16; off >= 1; off >>= 1) sm += __shfl_xor(sm, off, 32);
    p /= sm;
    probs[(size_t)t * NEXP + tid] = p;

    bool taken = false;
    float wsum = 0.f;
    float tw[TOPK];
    int ti[TOPK];
#pragma unroll
    for (int k = 0; k < TOPK; ++k) {
      float val = taken ? -1e30f : p;
      int idx = tid;
#pragma unroll
      for (int off = 16; off >= 1; off >>= 1) {
        float ov = __shfl_xor(val, off, 32);
        int oi = __shfl_xor(idx, off, 32);
        if (ov > val || (ov == val && oi < idx)) { val = ov; idx = oi; }
      }
      if (tid == idx) taken = true;
      ti[k] = idx;
      tw[k] = val;
      wsum += val;
    }
    if (tid == 0) {
      for (int k = 0; k < TOPK; ++k) {
        topi[t * TOPK + k] = ti[k];
        topw[t * TOPK + k] = tw[k] / wsum;
      }
    }
  }
}

__global__ void zero_kernel(int* p, int n) {
  int i = blockIdx.x * blockDim.x + threadIdx.x;
  if (i < n) p[i] = 0;
}

__global__ __launch_bounds__(256)
void count_kernel(const int* __restrict__ topi, int* __restrict__ counts) {
  int i = blockIdx.x * 256 + threadIdx.x;
  if (i < T_TOK * TOPK) atomicAdd(&counts[topi[i]], 1);
}

__global__ void scan_kernel(const int* __restrict__ counts, int* __restrict__ offsets) {
  if (threadIdx.x == 0) {
    int s = 0;
    for (int e = 0; e < NEXP; ++e) { offsets[e] = s; s += counts[e]; }
    offsets[NEXP] = s;
  }
}

__global__ __launch_bounds__(256)
void slot_kernel(const int* __restrict__ topi, const int* __restrict__ offsets,
                 int* __restrict__ cursor, int* __restrict__ token_list,
                 int* __restrict__ slot_of) {
  int i = blockIdx.x * 256 + threadIdx.x;
  if (i >= T_TOK * TOPK) return;
  int e = topi[i];
  int slot = offsets[e] + atomicAdd(&cursor[e], 1);
  token_list[slot] = i >> 2;
  slot_of[i] = slot;
}

__global__ __launch_bounds__(32)
void aux_kernel(const float* __restrict__ probs, const int* __restrict__ topi,
                float* __restrict__ out_aux) {
  int e = threadIdx.x;
  float imp = 0.f;
  for (int t = 0; t < T_TOK; ++t) imp += probs[(size_t)t * NEXP + e];
  int cnt = 0;
  for (int i = 0; i < T_TOK * TOPK; ++i) cnt += (topi[i] == e) ? 1 : 0;
  float a = (imp / (float)T_TOK) * ((float)cnt / (float)T_TOK);
#pragma unroll
  for (int off = 16; off >= 1; off >>= 1) a += __shfl_xor(a, off, 32);
  if (e == 0) out_aux[0] = (float)NEXP * a;
}

__global__ __launch_bounds__(256)
void combine_kernel(const float* __restrict__ x1, const float* __restrict__ sho,
                    const float* __restrict__ ys, const float* __restrict__ topw,
                    const int* __restrict__ slot_of, float* __restrict__ out) {
  int idx = blockIdx.x * 256 + threadIdx.x;
  int t = idx >> 9;
  int d = idx & 511;
  float acc = x1[idx] + sho[idx];
#pragma unroll
  for (int k = 0; k < TOPK; ++k) {
    acc += topw[t * TOPK + k] * ys[(size_t)slot_of[t * TOPK + k] * DMODEL + d];
  }
  out[idx] = acc;
}

// ---------------------------------------------------------------------------
extern "C" void kernel_launch(void* const* d_in, const int* in_sizes, int n_in,
                              void* d_out, int out_size, void* d_ws, size_t ws_size,
                              hipStream_t stream) {
  (void)in_sizes; (void)n_in; (void)ws_size;
  const float* x    = (const float*)d_in[0];
  const float* ln1g = (const float*)d_in[1];
  const float* ln1b = (const float*)d_in[2];
  const float* wq   = (const float*)d_in[3];
  const float* bq   = (const float*)d_in[4];
  const float* wk   = (const float*)d_in[5];
  const float* bk   = (const float*)d_in[6];
  const float* wv   = (const float*)d_in[7];
  const float* bv   = (const float*)d_in[8];
  const float* wo   = (const float*)d_in[9];
  const float* bo   = (const float*)d_in[10];
  const float* ln2g = (const float*)d_in[11];
  const float* ln2b = (const float*)d_in[12];
  const float* ra   = (const float*)d_in[13];
  const float* rb   = (const float*)d_in[14];
  const float* rew1 = (const float*)d_in[15];
  const float* reb1 = (const float*)d_in[16];
  const float* rew2 = (const float*)d_in[17];
  const float* reb2 = (const float*)d_in[18];
  const float* sew1 = (const float*)d_in[19];
  const float* seb1 = (const float*)d_in[20];
  const float* sew2 = (const float*)d_in[21];
  const float* seb2 = (const float*)d_in[22];

  char* base = (char*)d_ws;
  size_t off = 0;
  auto alloc = [&](size_t bytes) -> void* {
    void* p = base + off;
    off += (bytes + 255) & ~(size_t)255;
    return p;
  };

  __bf16* h1   = (__bf16*)alloc((size_t)T_TOK * DMODEL * 2);
  __bf16* qb_  = (__bf16*)alloc((size_t)T_TOK * DMODEL * 2);
  __bf16* kTb  = (__bf16*)alloc((size_t)T_TOK * DMODEL * 2);
  __bf16* vb   = (__bf16*)alloc((size_t)T_TOK * DMODEL * 2);
  __bf16* attb = (__bf16*)alloc((size_t)T_TOK * DMODEL * 2);
  float*  x1   = (float*) alloc((size_t)T_TOK * DMODEL * 4);
  __bf16* h2   = (__bf16*)alloc((size_t)T_TOK * DMODEL * 2);
  float*  probs= (float*) alloc((size_t)T_TOK * NEXP * 4);
  int*    topi = (int*)   alloc((size_t)T_TOK * TOPK * 4);
  float*  topw = (float*) alloc((size_t)T_TOK * TOPK * 4);
  int*    cc   = (int*)   alloc(64 * 4);          // counts[32] + cursor[32]
  int*    offs = (int*)   alloc(33 * 4);
  int*    tlist= (int*)   alloc((size_t)T_TOK * TOPK * 4);
  int*    sof  = (int*)   alloc((size_t)T_TOK * TOPK * 4);
  __bf16* hid  = (__bf16*)alloc((size_t)T_TOK * TOPK * DMODEL * 2);
  float*  ys   = (float*) alloc((size_t)T_TOK * TOPK * DMODEL * 4);
  __bf16* shh  = (__bf16*)alloc((size_t)T_TOK * 4096 * 2);
  float*  sho  = (float*) alloc((size_t)T_TOK * DMODEL * 4);
  // bf16 weight copies (L2-resident after first touch)
  __bf16* wqB   = (__bf16*)alloc((size_t)512 * 512 * 2);
  __bf16* wkB   = (__bf16*)alloc((size_t)512 * 512 * 2);
  __bf16* wvB   = (__bf16*)alloc((size_t)512 * 512 * 2);
  __bf16* woB   = (__bf16*)alloc((size_t)512 * 512 * 2);
  __bf16* sew1B = (__bf16*)alloc((size_t)512 * 4096 * 2);
  __bf16* sew2B = (__bf16*)alloc((size_t)4096 * 512 * 2);
  __bf16* rew1B = (__bf16*)alloc((size_t)NEXP * 512 * 512 * 2);
  __bf16* rew2B = (__bf16*)alloc((size_t)NEXP * 512 * 512 * 2);

  int* counts = cc;
  int* cursor = cc + 32;

  // weight conversion
  cvt_bf16_kernel<<<128, 256, 0, stream>>>(wq, wqB, 512 * 512);
  cvt_bf16_kernel<<<128, 256, 0, stream>>>(wk, wkB, 512 * 512);
  cvt_bf16_kernel<<<128, 256, 0, stream>>>(wv, wvB, 512 * 512);
  cvt_bf16_kernel<<<128, 256, 0, stream>>>(wo, woB, 512 * 512);
  cvt_bf16_kernel<<<1024, 256, 0, stream>>>(sew1, sew1B, 512 * 4096);
  cvt_bf16_kernel<<<1024, 256, 0, stream>>>(sew2, sew2B, 4096 * 512);
  cvt_bf16_kernel<<<4096, 256, 0, stream>>>(rew1, rew1B, NEXP * 512 * 512);
  cvt_bf16_kernel<<<4096, 256, 0, stream>>>(rew2, rew2B, NEXP * 512 * 512);

  zero_kernel<<<1, 64, 0, stream>>>(cc, 64);

  ln_kernel<<<T_TOK, 256, 0, stream>>>(x, ln1g, ln1b, h1);

  dim3 g512(32, 8);
  gemm_dense<0><<<g512, 256, 0, stream>>>(h1, wqB, bq, nullptr, qb_, nullptr, T_TOK, 512, 512);
  gemm_dense<1><<<g512, 256, 0, stream>>>(h1, wkB, bk, nullptr, kTb, nullptr, T_TOK, 512, 512);
  gemm_dense<0><<<g512, 256, 0, stream>>>(h1, wvB, bv, nullptr, vb, nullptr, T_TOK, 512, 512);

  attn_kernel<<<2048, 32, 0, stream>>>(qb_, kTb, vb, attb);

  gemm_dense<3><<<g512, 256, 0, stream>>>(attb, woB, bo, x, nullptr, x1, T_TOK, 512, 512);

  ln_kernel<<<T_TOK, 256, 0, stream>>>(x1, ln2g, ln2b, h2);

  router_kernel<<<T_TOK, 64, 0, stream>>>(h2, ra, rb, probs, topi, topw);
  count_kernel<<<64, 256, 0, stream>>>(topi, counts);
  scan_kernel<<<1, 32, 0, stream>>>(counts, offs);
  slot_kernel<<<64, 256, 0, stream>>>(topi, offs, cursor, tlist, sof);

  gemm_expert<true><<<dim3(128, 8, 32), 256, 0, stream>>>(h2, rew1B, reb1, hid,
                                                          nullptr, tlist, offs);
  gemm_expert<false><<<dim3(128, 8, 32), 256, 0, stream>>>(hid, rew2B, reb2, nullptr,
                                                           ys, tlist, offs);

  gemm_dense<2><<<dim3(32, 64), 256, 0, stream>>>(h2, sew1B, seb1, nullptr, shh,
                                                  nullptr, T_TOK, 4096, 512);
  gemm_dense<4><<<g512, 256, 0, stream>>>(shh, sew2B, seb2, nullptr, nullptr, sho,
                                          T_TOK, 512, 4096);

  aux_kernel<<<1, 32, 0, stream>>>(probs, topi, (float*)d_out + (out_size - 1));
  combine_kernel<<<(T_TOK * DMODEL) / 256, 256, 0, stream>>>(x1, sho, ys, topw, sof,
                                                             (float*)d_out);
}
